// SplashAttentionQKNormWrapper_69071664054923
// MI455X (gfx1250) — matrix operand order repaired and checked by reading, not compiled, not run on hardware
//
#include <hip/hip_runtime.h>
#include <hip/hip_bf16.h>

typedef __attribute__((ext_vector_type(16))) _Float16 v16h;
typedef __attribute__((ext_vector_type(8)))  float    v8f;

#define NHEAD 16
#define NKVH  8
#define DH    128
#define SEQ   2048
#define BATCH 2
#define HIDDEN 2048
#define MTOK  (BATCH * SEQ)      // 4096
#define NQKV  4096               // NH*D + 2*NKV*D

// ---------------------------------------------------------------------------
// CDNA5 async global->LDS copy (ASYNCcnt-tracked, no VGPR round trip).
// VDST holds the LDS byte address; flat LDS addresses map to LDS offsets via
// their low 32 bits (ISA 10.2 aperture rules), so truncation is sufficient.
// ---------------------------------------------------------------------------
__device__ __forceinline__ unsigned lds_addr(const void* p) {
  return (unsigned)(uintptr_t)p;
}
__device__ __forceinline__ void async_b128(unsigned lds_off, const void* g) {
  asm volatile("global_load_async_to_lds_b128 %0, %1, off"
               :: "v"(lds_off), "v"(g) : "memory");
}
__device__ __forceinline__ void wait_async0() {
  asm volatile("s_wait_asynccnt 0x0" ::: "memory");
}

// ---------------------------------------------------------------------------
// fp32 -> f16 conversion (row-major copy)
// ---------------------------------------------------------------------------
__global__ void cvt_f32_f16(const float* __restrict__ s, _Float16* __restrict__ d, int n) {
  int i = blockIdx.x * blockDim.x + threadIdx.x;
  int stride = gridDim.x * blockDim.x;
  for (; i < n; i += stride) d[i] = (_Float16)s[i];
}

// pack Wq|Wk|Wv TRANSPOSED into one f16 matrix [4096][2048]  (wqkvT[n][k])
__global__ void pack_wqkv_t(const float* __restrict__ Wq, const float* __restrict__ Wk,
                            const float* __restrict__ Wv, _Float16* __restrict__ o) {
  int i = blockIdx.x * blockDim.x + threadIdx.x;
  int stride = gridDim.x * blockDim.x;
  const int n = NQKV * HIDDEN;
  for (; i < n; i += stride) {
    int nn = i >> 11;         // output row    (0..4095)
    int k  = i & 2047;        // output col = K index
    float v;
    if (nn < 2048)      v = Wq[(size_t)k * 2048 + nn];
    else if (nn < 3072) v = Wk[(size_t)k * 1024 + (nn - 2048)];
    else                v = Wv[(size_t)k * 1024 + (nn - 3072)];
    o[i] = (_Float16)v;
  }
}

// Wo TRANSPOSED -> f16 [2048][2048]  (woT[n][k])
__global__ void cvt_wo_t(const float* __restrict__ Wo, _Float16* __restrict__ o) {
  int i = blockIdx.x * blockDim.x + threadIdx.x;
  int stride = gridDim.x * blockDim.x;
  const int n = HIDDEN * HIDDEN;
  for (; i < n; i += stride) {
    int nn = i >> 11, k = i & 2047;
    o[i] = (_Float16)Wo[(size_t)k * 2048 + nn];
  }
}

// ---------------------------------------------------------------------------
// WMMA GEMM: C(f32, M x N) = A(f16, M x K) * Bt(f16, N x K)^T
// Both A and Bt are row-major with row stride K. Block tile 128x128,
// 128 threads = 4 waves (2 x 2), wave tile 64x64 -> 4x4 16x16 accumulators
// (16 WMMAs per 16 ds_load_b128 per K-step -> 1:1 fragment-load:WMMA ratio).
// K staged 32 wide, double-buffered in LDS, filled with async global->LDS.
// ---------------------------------------------------------------------------
__global__ __launch_bounds__(128)
void gemm_f16f32(const _Float16* __restrict__ A, const _Float16* __restrict__ Bt,
                 float* __restrict__ C, int ldc, int K) {
  __shared__ _Float16 Al[2][128][48];   // 12 KiB per buffer
  __shared__ _Float16 Bl[2][128][48];

  const int tid  = threadIdx.x;
  const int lane = tid & 31;
  const int wid  = tid >> 5;
  const int wm   = wid & 1;
  const int wn   = wid >> 1;
  const int hl   = lane >> 4;
  const int l16  = lane & 15;
  const int bM   = blockIdx.y * 128;
  const int bN   = blockIdx.x * 128;

  // staging: 4096 f16 per tile = 512 b128 chunks, 4 per thread per operand
  const unsigned aBase = lds_addr(&Al[0][0][0]);
  const unsigned bBase = lds_addr(&Bl[0][0][0]);
  const unsigned bufB  = 128 * 48 * 2;   // byte stride between double buffers

  v8f acc[4][4] = {};

  // prologue: tile 0 into buffer 0
#pragma unroll
  for (int it = 0; it < 4; ++it) {
    int e = (it * 128 + tid) * 8;
    int r = e >> 5, c = e & 31;
    async_b128(aBase + (unsigned)(r * 48 + c) * 2, A  + (size_t)(bM + r) * K + c);
    async_b128(bBase + (unsigned)(r * 48 + c) * 2, Bt + (size_t)(bN + r) * K + c);
  }
  wait_async0();
  __syncthreads();

  int p = 0;
  for (int k0 = 0; k0 < K; k0 += 32) {
    // issue async loads for the next K tile into the other buffer
    if (k0 + 32 < K) {
      unsigned d = (unsigned)(p ^ 1) * bufB;
      int kn = k0 + 32;
#pragma unroll
      for (int it = 0; it < 4; ++it) {
        int e = (it * 128 + tid) * 8;
        int r = e >> 5, c = e & 31;
        async_b128(aBase + d + (unsigned)(r * 48 + c) * 2,
                   A  + (size_t)(bM + r) * K + kn + c);
        async_b128(bBase + d + (unsigned)(r * 48 + c) * 2,
                   Bt + (size_t)(bN + r) * K + kn + c);
      }
    }

    // compute on current buffer: 16 WMMAs
    v16h afr[4];
#pragma unroll
    for (int mt = 0; mt < 4; ++mt) {
      const _Float16* ap = &Al[p][wm * 64 + mt * 16 + l16][hl * 8];
      ((uint4*)&afr[mt])[0] = *(const uint4*)ap;         // K = hl*8 + 0..7
      ((uint4*)&afr[mt])[1] = *(const uint4*)(ap + 16);  // K = 16 + hl*8 + 0..7
    }
#pragma unroll
    for (int nt = 0; nt < 4; ++nt) {
      v16h bfr;
      const _Float16* bp = &Bl[p][wn * 64 + nt * 16 + l16][hl * 8];
      ((uint4*)&bfr)[0] = *(const uint4*)bp;             // K = hl*8 + 0..7
      ((uint4*)&bfr)[1] = *(const uint4*)(bp + 16);      // K = 16 + hl*8 + 0..7
#pragma unroll
      for (int mt = 0; mt < 4; ++mt)
        acc[mt][nt] = __builtin_amdgcn_wmma_f32_16x16x32_f16(
            false, afr[mt], false, bfr, (short)0, acc[mt][nt], false, false);
    }

    wait_async0();
    __syncthreads();
    p ^= 1;
  }

#pragma unroll
  for (int mt = 0; mt < 4; ++mt)
#pragma unroll
    for (int nt = 0; nt < 4; ++nt) {
      int col = bN + wn * 64 + nt * 16 + l16;
#pragma unroll
      for (int r = 0; r < 8; ++r) {
        int row = bM + wm * 64 + mt * 16 + r + hl * 8;
        C[(size_t)row * ldc + col] = acc[mt][nt][r];
      }
    }
}

// ---------------------------------------------------------------------------
// Fused RMS-norm + RoPE + q-scale. V is written TRANSPOSED per kv head:
// vT[b][kvh][d][s], so the attention kernel can async-stage it directly.
// ---------------------------------------------------------------------------
__global__ __launch_bounds__(256)
void rmsnorm_rope(const float* __restrict__ qkv,                // [MTOK][4096]
                  const float* __restrict__ cosb, const float* __restrict__ sinb,
                  const float* __restrict__ qw, const float* __restrict__ kw,
                  _Float16* __restrict__ qo,    // [B][NH][S][D]
                  _Float16* __restrict__ ko,    // [B][NKV][S][D]
                  _Float16* __restrict__ vt) {  // [B][NKV][D][S]
  const int tok  = blockIdx.x;
  const int b    = tok >> 11;
  const int s    = tok & 2047;
  const int lane = threadIdx.x & 31;
  const int wid  = threadIdx.x >> 5;
  const int d0   = lane * 4;

#pragma unroll
  for (int rr = 0; rr < 4; ++rr) {
    int idx = wid * 4 + rr;                 // 0..31 (16 q, 8 k, 8 v)
    int cls, h, coloff;
    if (idx < NHEAD)             { cls = 0; h = idx;                coloff = h * DH; }
    else if (idx < NHEAD + NKVH) { cls = 1; h = idx - NHEAD;        coloff = 2048 + h * DH; }
    else                         { cls = 2; h = idx - NHEAD - NKVH; coloff = 3072 + h * DH; }

    const float* src = qkv + (size_t)tok * NQKV + coloff;
    float4 x = *(const float4*)(src + d0);

    if (cls == 2) {
      _Float16* dst = vt + ((size_t)(b * NKVH + h) * DH + d0) * SEQ + s;
      dst[0 * SEQ] = (_Float16)x.x; dst[1 * SEQ] = (_Float16)x.y;
      dst[2 * SEQ] = (_Float16)x.z; dst[3 * SEQ] = (_Float16)x.w;
      continue;
    }

    // RMS norm over D=128
    float ss = x.x * x.x + x.y * x.y + x.z * x.z + x.w * x.w;
#pragma unroll
    for (int m = 16; m >= 1; m >>= 1) ss += __shfl_xor(ss, m, 32);
    float inv = rsqrtf(ss * (1.0f / DH) + 1e-6f);

    const float* w = (cls == 0) ? qw : kw;
    float4 wv = *(const float4*)(w + d0);
    float n0 = x.x * inv * wv.x, n1 = x.y * inv * wv.y;
    float n2 = x.z * inv * wv.z, n3 = x.w * inv * wv.w;

    // RoPE: partner element lives 16 lanes away (d +/- 64)
    const float* cp = cosb + ((size_t)b * SEQ + s) * DH + d0;
    const float* sp = sinb + ((size_t)b * SEQ + s) * DH + d0;
    float4 c  = *(const float4*)cp;
    float4 sn = *(const float4*)sp;
    float p0 = __shfl_xor(n0, 16, 32);
    float p1 = __shfl_xor(n1, 16, 32);
    float p2 = __shfl_xor(n2, 16, 32);
    float p3 = __shfl_xor(n3, 16, 32);
    float sgn = (lane < 16) ? -1.0f : 1.0f;
    float r0 = n0 * c.x + sgn * p0 * sn.x;
    float r1 = n1 * c.y + sgn * p1 * sn.y;
    float r2 = n2 * c.z + sgn * p2 * sn.z;
    float r3 = n3 * c.w + sgn * p3 * sn.w;

    float scale = (cls == 0) ? 0.08838834764831845f : 1.0f;  // D^-0.5 folded into q
    _Float16* dst = (cls == 0)
        ? qo + ((size_t)(b * NHEAD + h) * SEQ + s) * DH + d0
        : ko + ((size_t)(b * NKVH + h) * SEQ + s) * DH + d0;
    dst[0] = (_Float16)(r0 * scale); dst[1] = (_Float16)(r1 * scale);
    dst[2] = (_Float16)(r2 * scale); dst[3] = (_Float16)(r3 * scale);
  }
}

// ---------------------------------------------------------------------------
// Causal flash attention. grid = (S/64, B*NH); 128 threads = 4 waves,
// each wave owns 16 q rows. All tile staging is async global->LDS.
// ---------------------------------------------------------------------------
__global__ __launch_bounds__(128)
void flash_attn(const _Float16* __restrict__ Q, const _Float16* __restrict__ Kc,
                const _Float16* __restrict__ Vt, _Float16* __restrict__ AO) {
  __shared__ _Float16 Qs[64][136];      // Q tile row-major
  __shared__ _Float16 Ks[64][136];      // K tile row-major
  __shared__ _Float16 Vl[128][72];      // V tile, already transposed: Vl[d][kpos]
  __shared__ _Float16 Ps[4][16][72];    // per-wave P strips

  const int tid  = threadIdx.x;
  const int lane = tid & 31;
  const int w    = tid >> 5;
  const int hl   = lane >> 4;
  const int l16  = lane & 15;
  const int qb   = blockIdx.x;
  const int bh   = blockIdx.y;
  const int b    = bh / NHEAD;
  const int h    = bh % NHEAD;
  const int kvh  = h >> 1;              // GQA rep = 2

  const _Float16* Qbase = Q  + ((size_t)(b * NHEAD + h) * SEQ + qb * 64) * DH;
  const _Float16* Kbase = Kc + ((size_t)(b * NKVH + kvh) * SEQ) * DH;
  const _Float16* Vbase = Vt + ((size_t)(b * NKVH + kvh) * DH) * SEQ;

  // stage Q once (async)
#pragma unroll
  for (int it = 0; it < 8; ++it) {
    int e = (it * 128 + tid) * 8;
    int r = e >> 7, c = e & 127;
    async_b128(lds_addr(&Qs[r][c]), Qbase + (size_t)r * DH + c);
  }
  wait_async0();
  __syncthreads();

  // cache Q A-fragments (4 K-chunks of 32 over D=128)
  v16h qfr[4];
#pragma unroll
  for (int dc = 0; dc < 4; ++dc) {
    const _Float16* p = &Qs[w * 16 + l16][dc * 32 + hl * 8];
    ((uint4*)&qfr[dc])[0] = *(const uint4*)p;
    ((uint4*)&qfr[dc])[1] = *(const uint4*)(p + 16);
  }

  v8f ao[8] = {};
  float mrow[8], lrow[8];
#pragma unroll
  for (int r = 0; r < 8; ++r) { mrow[r] = -3.0e38f; lrow[r] = 0.0f; }

  for (int kb = 0; kb <= qb; ++kb) {
    __syncthreads();   // previous tile fully consumed before overwrite
    // K tile: 64 rows x 128, row-major
#pragma unroll
    for (int it = 0; it < 8; ++it) {
      int e = (it * 128 + tid) * 8;
      int r = e >> 7, c = e & 127;
      async_b128(lds_addr(&Ks[r][c]),
                 Kbase + (size_t)(kb * 64 + r) * DH + c);
    }
    // V tile: 128 d-rows x 64 s-cols, source already transposed
#pragma unroll
    for (int it = 0; it < 8; ++it) {
      int e = (it * 128 + tid) * 8;
      int r = e >> 6, c = e & 63;
      async_b128(lds_addr(&Vl[r][c]),
                 Vbase + (size_t)r * SEQ + kb * 64 + c);
    }
    wait_async0();
    __syncthreads();

    // S strip = q(16 x 128) * K^T(128 x 64) -> 4 tiles of 16x16
    v8f st[4] = {};
#pragma unroll
    for (int t = 0; t < 4; ++t) {
#pragma unroll
      for (int dc = 0; dc < 4; ++dc) {
        v16h kfr;
        const _Float16* p = &Ks[t * 16 + l16][dc * 32 + hl * 8];
        ((uint4*)&kfr)[0] = *(const uint4*)p;
        ((uint4*)&kfr)[1] = *(const uint4*)(p + 16);
        st[t] = __builtin_amdgcn_wmma_f32_16x16x32_f16(
            false, qfr[dc], false, kfr, (short)0, st[t], false, false);
      }
    }

    // causal mask on the diagonal block
    if (kb == qb) {
#pragma unroll
      for (int t = 0; t < 4; ++t)
#pragma unroll
        for (int r = 0; r < 8; ++r) {
          int qrow = w * 16 + r + hl * 8;
          int kcol = t * 16 + l16;
          if (kcol > qrow) st[t][r] = -1.0e30f;
        }
    }

    // online softmax (row stats replicated across each 16-lane half)
    float alpha[8];
#pragma unroll
    for (int r = 0; r < 8; ++r) {
      float mx = fmaxf(fmaxf(st[0][r], st[1][r]), fmaxf(st[2][r], st[3][r]));
#pragma unroll
      for (int m = 8; m >= 1; m >>= 1) mx = fmaxf(mx, __shfl_xor(mx, m, 32));
      float mn = fmaxf(mrow[r], mx);
      alpha[r] = __expf(mrow[r] - mn);
      mrow[r] = mn;
    }
    float rsum[8];
#pragma unroll
    for (int r = 0; r < 8; ++r) rsum[r] = 0.0f;
#pragma unroll
    for (int t = 0; t < 4; ++t)
#pragma unroll
      for (int r = 0; r < 8; ++r) {
        float pv = __expf(st[t][r] - mrow[r]);
        st[t][r] = pv;
        rsum[r] += pv;
      }
#pragma unroll
    for (int r = 0; r < 8; ++r) {
#pragma unroll
      for (int m = 8; m >= 1; m >>= 1) rsum[r] += __shfl_xor(rsum[r], m, 32);
      lrow[r] = lrow[r] * alpha[r] + rsum[r];
    }
#pragma unroll
    for (int n = 0; n < 8; ++n)
#pragma unroll
      for (int r = 0; r < 8; ++r) ao[n][r] *= alpha[r];

    // P (C layout) -> LDS -> A-fragment layout (per-wave region, same-wave RAW)
#pragma unroll
    for (int t = 0; t < 4; ++t)
#pragma unroll
      for (int r = 0; r < 8; ++r)
        Ps[w][r + hl * 8][t * 16 + l16] = (_Float16)st[t][r];
    asm volatile("s_wait_dscnt 0" ::: "memory");

    v16h pfr[2];
#pragma unroll
    for (int kc = 0; kc < 2; ++kc) {
      const _Float16* p = &Ps[w][l16][kc * 32 + hl * 8];
      ((uint4*)&pfr[kc])[0] = *(const uint4*)p;
      ((uint4*)&pfr[kc])[1] = *(const uint4*)(p + 16);
    }

    // O(16 x 128) += P(16 x 64) * V(64 x 128)
#pragma unroll
    for (int n = 0; n < 8; ++n) {
#pragma unroll
      for (int kc = 0; kc < 2; ++kc) {
        v16h vfr;
        const _Float16* p = &Vl[n * 16 + l16][kc * 32 + hl * 8];
        ((uint4*)&vfr)[0] = *(const uint4*)p;
        ((uint4*)&vfr)[1] = *(const uint4*)(p + 16);
        ao[n] = __builtin_amdgcn_wmma_f32_16x16x32_f16(
            false, pfr[kc], false, vfr, (short)0, ao[n], false, false);
      }
    }
  }

  // epilogue: normalize, emit (b, s, h*D + d) f16 for the output projection
#pragma unroll
  for (int n = 0; n < 8; ++n)
#pragma unroll
    for (int r = 0; r < 8; ++r) {
      int srow = qb * 64 + w * 16 + r + hl * 8;
      int col  = h * DH + n * 16 + l16;
      AO[((size_t)b * SEQ + srow) * (NHEAD * DH) + col] = (_Float16)(ao[n][r] / lrow[r]);
    }
}

// ---------------------------------------------------------------------------
// launch
// ---------------------------------------------------------------------------
extern "C" void kernel_launch(void* const* d_in, const int* in_sizes, int n_in,
                              void* d_out, int out_size, void* d_ws, size_t ws_size,
                              hipStream_t stream) {
  const float* hs   = (const float*)d_in[0];
  const float* cosb = (const float*)d_in[1];
  const float* sinb = (const float*)d_in[2];
  const float* Wq   = (const float*)d_in[3];
  const float* Wk   = (const float*)d_in[4];
  const float* Wv   = (const float*)d_in[5];
  const float* Wo   = (const float*)d_in[6];
  const float* qw   = (const float*)d_in[7];
  const float* kw   = (const float*)d_in[8];
  float* out = (float*)d_out;

  char* ws = (char*)d_ws;
  size_t off = 0;
  auto alloc = [&](size_t bytes) -> void* {
    void* p = ws + off;
    off = (off + bytes + 255) & ~(size_t)255;
    return p;
  };

  _Float16* hsF   = (_Float16*)alloc((size_t)MTOK * HIDDEN * 2);     // 16 MiB
  _Float16* wqkvT = (_Float16*)alloc((size_t)NQKV * HIDDEN * 2);     // 16 MiB
  _Float16* woT   = (_Float16*)alloc((size_t)HIDDEN * HIDDEN * 2);   //  8 MiB
  float*    qkvF  = (float*)   alloc((size_t)MTOK * NQKV * 4);       // 64 MiB
  _Float16* qF    = (_Float16*)alloc((size_t)BATCH * NHEAD * SEQ * DH * 2); // 16 MiB
  _Float16* kF    = (_Float16*)alloc((size_t)BATCH * NKVH * SEQ * DH * 2);  //  8 MiB
  _Float16* vTF   = (_Float16*)alloc((size_t)BATCH * NKVH * DH * SEQ * 2);  //  8 MiB
  _Float16* aoF   = (_Float16*)alloc((size_t)MTOK * HIDDEN * 2);     // 16 MiB

  cvt_f32_f16<<<1024, 256, 0, stream>>>(hs, hsF, MTOK * HIDDEN);
  pack_wqkv_t<<<1024, 256, 0, stream>>>(Wq, Wk, Wv, wqkvT);
  cvt_wo_t   <<<512, 256, 0, stream>>>(Wo, woT);

  // QKV projection: [4096 x 2048] x [2048 x 4096] -> fp32
  gemm_f16f32<<<dim3(NQKV / 128, MTOK / 128), 128, 0, stream>>>(
      hsF, wqkvT, qkvF, NQKV, HIDDEN);

  rmsnorm_rope<<<MTOK, 256, 0, stream>>>(qkvF, cosb, sinb, qw, kw, qF, kF, vTF);

  flash_attn<<<dim3(SEQ / 64, BATCH * NHEAD), 128, 0, stream>>>(qF, kF, vTF, aoF);

  // Output projection: [4096 x 2048] x [2048 x 2048] -> fp32 out
  gemm_f16f32<<<dim3(HIDDEN / 128, MTOK / 128), 128, 0, stream>>>(
      aoF, woT, out, HIDDEN, HIDDEN);
}
